// VAEGuidedAttention_44023414784303
// MI455X (gfx1250) — matrix operand order repaired and checked by reading, not compiled, not run on hardware
//
#include <hip/hip_runtime.h>
#include <hip/hip_bf16.h>

typedef __attribute__((ext_vector_type(8)))  __bf16 v8bf;
typedef __attribute__((ext_vector_type(16))) __bf16 v16bf;
typedef __attribute__((ext_vector_type(8)))  float  v8f;

#define V8F_ZERO {0.f,0.f,0.f,0.f,0.f,0.f,0.f,0.f}

__device__ inline v16bf cat16(v8bf lo, v8bf hi) {
  return __builtin_shufflevector(lo, hi, 0,1,2,3,4,5,6,7,8,9,10,11,12,13,14,15);
}

// A-matrix 16x32 (MxK) bf16 fragment. Row pointer p = &A[m][k0], m = lane%16.
// Lanes 0-15: halves = K {0..7, 16..23}; lanes 16-31: K {8..15, 24..31}.
__device__ inline v16bf load_a_frag(const __bf16* p, int lane) {
  const int b1 = (lane & 16) ? 8 : 0;
  v8bf lo = *(const v8bf*)(p + b1);
  v8bf hi = *(const v8bf*)(p + b1 + 16);
  return cat16(lo, hi);
}

// B-matrix 32x16 (KxN) bf16 fragment, sourced from k-contiguous storage
// (B[k][n] = row_n[k]). p = &row_n[k0], n = lane%16.
// Lanes 0-15: halves = K {0..7, 16..23}; lanes 16-31: K {8..15, 24..31}.
__device__ inline v16bf load_b_frag(const __bf16* p, int lane) {
  const int kb = (lane & 16) ? 8 : 0;
  v8bf lo = *(const v8bf*)(p + kb);
  v8bf hi = *(const v8bf*)(p + kb + 16);
  return cat16(lo, hi);
}

__device__ inline v8f wmma_bf16(v16bf a, v16bf b, v8f c) {
  return __builtin_amdgcn_wmma_f32_16x16x32_bf16(false, a, false, b, (short)0, c, false, false);
}

// ---------------- fp32 -> bf16 conversion ----------------
__global__ void cvt_bf16(const float* __restrict__ src, __bf16* __restrict__ dst, int n) {
  int i = blockIdx.x * blockDim.x + threadIdx.x;
  if (i < n) dst[i] = (__bf16)src[i];
}

// ---------------- mean over sequence ----------------
__global__ void mean_pool(const float* __restrict__ x, float* __restrict__ xm) {
  int idx = blockIdx.x * blockDim.x + threadIdx.x;   // 0..1023
  int b = idx >> 9, d = idx & 511;
  float s = 0.f;
  for (int t = 0; t < 4096; ++t) s += x[((size_t)b * 4096 + t) * 512 + d];
  xm[idx] = s * (1.f / 4096.f);
}

// ---------------- tiny VAE MLP (1 block, 512 threads) ----------------
__global__ void vae_mlp(const float* xm, const float* eps,
                        const float* we1, const float* be1,
                        const float* we2, const float* be2,
                        const float* wmu, const float* bmu,
                        const float* wlv, const float* blv,
                        const float* wf,  const float* bfv,
                        float* mu_out, float* lv_out, float* lat) {
  __shared__ float h1[256];
  __shared__ float h2[128];
  __shared__ float zs[64];
  int t = threadIdx.x;
  for (int b = 0; b < 2; ++b) {
    if (t < 256) {
      float s = be1[t];
      for (int k = 0; k < 512; ++k) s += xm[b * 512 + k] * we1[t * 512 + k];
      h1[t] = fmaxf(s, 0.f);
    }
    __syncthreads();
    if (t < 128) {
      float s = be2[t];
      for (int k = 0; k < 256; ++k) s += h1[k] * we2[t * 256 + k];
      h2[t] = fmaxf(s, 0.f);
    }
    __syncthreads();
    if (t < 64) {
      float m = bmu[t], l = blv[t];
      for (int k = 0; k < 128; ++k) { m += h2[k] * wmu[t * 128 + k]; l += h2[k] * wlv[t * 128 + k]; }
      mu_out[b * 64 + t] = m;
      lv_out[b * 64 + t] = l;
      zs[t] = m + eps[b * 64 + t] * __expf(0.5f * l);
    }
    __syncthreads();
    {
      float s = bfv[t];
      for (int k = 0; k < 64; ++k) s += zs[k] * wf[t * 64 + k];
      lat[b * 512 + t] = s;
    }
    __syncthreads();
  }
}

// ---------------- QKV projection GEMM: 64x64 tile per wave ----------------
// C[M=8192, N=512] = X @ W^T (+ bias [+ latent for q/k]).
// which: 0=q -> qg[b,h,s,hd], 1=k -> kg[b,h,s,hd], 2=v -> vt[b,h,hd,s]
__global__ __launch_bounds__(128) void qkv_gemm(
    const __bf16* __restrict__ xb,
    const __bf16* __restrict__ wqb, const __bf16* __restrict__ wkb, const __bf16* __restrict__ wvb,
    const float* __restrict__ bq, const float* __restrict__ bk, const float* __restrict__ bv,
    const float* __restrict__ lat,
    __bf16* __restrict__ qg, __bf16* __restrict__ kg, __bf16* __restrict__ vt) {
  const int lane = threadIdx.x;
  const int l15 = lane & 15;
  const int cb = blockIdx.x * 4 + threadIdx.y;   // 0..7  : 64-col block
  const int tm = blockIdx.y;                     // 0..127: 64-row block
  const int which = blockIdx.z;
  const __bf16* W = (which == 0) ? wqb : (which == 1) ? wkb : wvb;
  const float* bias = (which == 0) ? bq : (which == 1) ? bk : bv;
  const __bf16* arow = xb + (size_t)(tm * 64 + l15) * 512;   // + mt*16*512 folds into imm
  const __bf16* brow = W + (size_t)(cb * 64 + l15) * 512;    // + nt*16*512 folds into imm
  v8f c[4][4];
#pragma unroll
  for (int mt = 0; mt < 4; ++mt)
#pragma unroll
    for (int nt = 0; nt < 4; ++nt) { v8f z = V8F_ZERO; c[mt][nt] = z; }
  for (int k0 = 0; k0 < 512; k0 += 32) {
    v16bf a[4], b[4];
#pragma unroll
    for (int mt = 0; mt < 4; ++mt) a[mt] = load_a_frag(arow + mt * (16 * 512) + k0, lane);
#pragma unroll
    for (int nt = 0; nt < 4; ++nt) b[nt] = load_b_frag(brow + nt * (16 * 512) + k0, lane);
#pragma unroll
    for (int mt = 0; mt < 4; ++mt)
#pragma unroll
      for (int nt = 0; nt < 4; ++nt) c[mt][nt] = wmma_bf16(a[mt], b[nt], c[mt][nt]);
  }
  const int bidx = (tm * 64) >> 12;
  const int sbase = ((tm * 64) & 4095) + ((lane & 16) ? 8 : 0);
#pragma unroll
  for (int nt = 0; nt < 4; ++nt) {
    const int col = cb * 64 + nt * 16 + l15;
    float badd = bias[col];
    if (which < 2) badd += lat[bidx * 512 + col];
    const int h = col >> 6, d = col & 63;
#pragma unroll
    for (int mt = 0; mt < 4; ++mt) {
#pragma unroll
      for (int r = 0; r < 8; ++r) {
        int s = sbase + mt * 16 + r;
        float val = c[mt][nt][r] + badd;
        if (which == 0)
          qg[(((size_t)bidx * 8 + h) * 4096 + s) * 64 + d] = (__bf16)val;
        else if (which == 1)
          kg[(((size_t)bidx * 8 + h) * 4096 + s) * 64 + d] = (__bf16)val;
        else
          vt[(((size_t)bidx * 8 + h) * 64 + d) * 4096 + s] = (__bf16)val;
      }
    }
  }
}

// ---------------- flash attention: 32 Q rows per wave, K/V shared ----------------
__global__ __launch_bounds__(128) void attn_fa(
    const __bf16* __restrict__ qg, const __bf16* __restrict__ kg,
    const __bf16* __restrict__ vt, __bf16* __restrict__ ctxb) {
  __shared__ __align__(16) __bf16 pt[4][2][16][32];  // per-wave double P tile
  const int lane = threadIdx.x & 31;
  const int wid  = threadIdx.x >> 5;
  const int gw   = blockIdx.x * 4 + wid;             // 0..2047 waves
  const int qt2  = gw & 127;                         // 32-row Q block within S
  const int bh   = gw >> 7;                          // 0..15 (b*H+h)
  const __bf16* Q = qg + (size_t)bh * 4096 * 64;
  const __bf16* K = kg + (size_t)bh * 4096 * 64;
  const __bf16* V = vt + (size_t)bh * 64 * 4096;
  const int l15 = lane & 15;
  const int q0 = qt2 * 32;
  const __bf16* qrow = Q + (size_t)(q0 + l15) * 64;  // + t*16*64 folds into imm
  v16bf aq[2][2];
#pragma unroll
  for (int t = 0; t < 2; ++t) {
    aq[t][0] = load_a_frag(qrow + t * (16 * 64), lane);       // d 0..31
    aq[t][1] = load_a_frag(qrow + t * (16 * 64) + 32, lane);  // d 32..63
  }
  v8f c[2][4];
#pragma unroll
  for (int t = 0; t < 2; ++t)
#pragma unroll
    for (int g = 0; g < 4; ++g) { v8f z = V8F_ZERO; c[t][g] = z; }
  float mrow[2][8], lrow[2][8];
#pragma unroll
  for (int t = 0; t < 2; ++t)
#pragma unroll
    for (int r = 0; r < 8; ++r) { mrow[t][r] = -1e30f; lrow[t][r] = 0.f; }

  const __bf16* vbase = V + (size_t)l15 * 4096;      // + g*16*4096 folds into imm

  for (int j0 = 0; j0 < 4096; j0 += 32) {
    const __bf16* kr0 = K + (size_t)(j0 + l15) * 64;
    // prefetch next key block (global_prefetch_b8; no counter cost)
    if (j0 + 32 < 4096) {
      __builtin_prefetch(kr0 + 32 * 64, 0, 3);
      __builtin_prefetch(kr0 + 48 * 64, 0, 3);
      __builtin_prefetch(vbase + j0 + 32, 0, 3);
    }
    // K fragments shared by both Q tiles: cols j (2 blocks) x d-window (2)
    v16bf kb00 = load_b_frag(kr0, lane);
    v16bf kb01 = load_b_frag(kr0 + 32, lane);
    v16bf kb10 = load_b_frag(kr0 + 16 * 64, lane);
    v16bf kb11 = load_b_frag(kr0 + 16 * 64 + 32, lane);
    __builtin_amdgcn_wave_barrier();
#pragma unroll
    for (int t = 0; t < 2; ++t) {
      v8f s0 = V8F_ZERO, s1 = V8F_ZERO;
      s0 = wmma_bf16(aq[t][0], kb00, s0);
      s0 = wmma_bf16(aq[t][1], kb01, s0);
      s1 = wmma_bf16(aq[t][0], kb10, s1);
      s1 = wmma_bf16(aq[t][1], kb11, s1);
      // online softmax: row stats reduced across each 16-lane half (C layout)
      v8f p0, p1;
      float alpha[8];
#pragma unroll
      for (int r = 0; r < 8; ++r) {
        float a0 = s0[r] * 0.125f, a1 = s1[r] * 0.125f;
        float mx = fmaxf(a0, a1);
        mx = fmaxf(mx, __shfl_xor(mx, 8, 32));
        mx = fmaxf(mx, __shfl_xor(mx, 4, 32));
        mx = fmaxf(mx, __shfl_xor(mx, 2, 32));
        mx = fmaxf(mx, __shfl_xor(mx, 1, 32));
        float mnew = fmaxf(mrow[t][r], mx);
        float al = __expf(mrow[t][r] - mnew);
        mrow[t][r] = mnew;
        float e0 = __expf(a0 - mnew);
        float e1 = __expf(a1 - mnew);
        float rs = e0 + e1;
        rs += __shfl_xor(rs, 8, 32);
        rs += __shfl_xor(rs, 4, 32);
        rs += __shfl_xor(rs, 2, 32);
        rs += __shfl_xor(rs, 1, 32);
        lrow[t][r] = lrow[t][r] * al + rs;
        alpha[r] = al;
        p0[r] = e0; p1[r] = e1;
      }
#pragma unroll
      for (int g = 0; g < 4; ++g)
#pragma unroll
        for (int r = 0; r < 8; ++r) c[t][g][r] *= alpha[r];
      // P (C layout) -> LDS tile t
#pragma unroll
      for (int r = 0; r < 8; ++r) {
        int row = ((lane & 16) ? 8 : 0) + r;
        pt[wid][t][row][l15]      = (__bf16)p0[r];
        pt[wid][t][row][16 + l15] = (__bf16)p1[r];
      }
    }
    asm volatile("s_wait_dscnt 0x0" ::: "memory");
    __builtin_amdgcn_wave_barrier();
    v16bf pf[2];
#pragma unroll
    for (int t = 0; t < 2; ++t) pf[t] = load_a_frag(&pt[wid][t][l15][0], lane);
    // V fragments shared by both Q tiles
    v16bf vb[4];
#pragma unroll
    for (int g = 0; g < 4; ++g) vb[g] = load_b_frag(vbase + g * (16 * 4096) + j0, lane);
#pragma unroll
    for (int t = 0; t < 2; ++t)
#pragma unroll
      for (int g = 0; g < 4; ++g) c[t][g] = wmma_bf16(pf[t], vb[g], c[t][g]);
    __builtin_amdgcn_wave_barrier();
  }
  const int b = bh >> 3, h = bh & 7;
#pragma unroll
  for (int t = 0; t < 2; ++t)
#pragma unroll
    for (int r = 0; r < 8; ++r) {
      float inv = 1.f / lrow[t][r];
      int s = q0 + t * 16 + ((lane & 16) ? 8 : 0) + r;
#pragma unroll
      for (int g = 0; g < 4; ++g)
        ctxb[((size_t)b * 4096 + s) * 512 + h * 64 + g * 16 + l15] = (__bf16)(c[t][g][r] * inv);
    }
}

// ---------------- output projection GEMM: 64x64 tile per wave, fp32 out ----------------
__global__ __launch_bounds__(128) void out_gemm(
    const __bf16* __restrict__ ctxb, const __bf16* __restrict__ wob,
    const float* __restrict__ bo, float* __restrict__ out) {
  const int lane = threadIdx.x;
  const int l15 = lane & 15;
  const int cb = blockIdx.x * 4 + threadIdx.y;   // 0..7
  const int tm = blockIdx.y;                     // 0..127
  const __bf16* arow = ctxb + (size_t)(tm * 64 + l15) * 512;
  const __bf16* brow = wob + (size_t)(cb * 64 + l15) * 512;
  v8f c[4][4];
#pragma unroll
  for (int mt = 0; mt < 4; ++mt)
#pragma unroll
    for (int nt = 0; nt < 4; ++nt) { v8f z = V8F_ZERO; c[mt][nt] = z; }
  for (int k0 = 0; k0 < 512; k0 += 32) {
    v16bf a[4], b[4];
#pragma unroll
    for (int mt = 0; mt < 4; ++mt) a[mt] = load_a_frag(arow + mt * (16 * 512) + k0, lane);
#pragma unroll
    for (int nt = 0; nt < 4; ++nt) b[nt] = load_b_frag(brow + nt * (16 * 512) + k0, lane);
#pragma unroll
    for (int mt = 0; mt < 4; ++mt)
#pragma unroll
      for (int nt = 0; nt < 4; ++nt) c[mt][nt] = wmma_bf16(a[mt], b[nt], c[mt][nt]);
  }
  const int rbase = tm * 64 + ((lane & 16) ? 8 : 0);
#pragma unroll
  for (int nt = 0; nt < 4; ++nt) {
    const int col = cb * 64 + nt * 16 + l15;
    const float badd = bo[col];
#pragma unroll
    for (int mt = 0; mt < 4; ++mt)
#pragma unroll
      for (int r = 0; r < 8; ++r)
        out[(size_t)(rbase + mt * 16 + r) * 512 + col] = c[mt][nt][r] + badd;
  }
}

extern "C" void kernel_launch(void* const* d_in, const int* in_sizes, int n_in,
                              void* d_out, int out_size, void* d_ws, size_t ws_size,
                              hipStream_t stream) {
  (void)in_sizes; (void)n_in; (void)out_size; (void)ws_size;
  const float* x   = (const float*)d_in[0];
  const float* eps = (const float*)d_in[1];
  const float* wq  = (const float*)d_in[2];
  const float* bq  = (const float*)d_in[3];
  const float* wk  = (const float*)d_in[4];
  const float* bk  = (const float*)d_in[5];
  const float* wv  = (const float*)d_in[6];
  const float* bv  = (const float*)d_in[7];
  const float* wo  = (const float*)d_in[8];
  const float* bo  = (const float*)d_in[9];
  const float* we1 = (const float*)d_in[10];
  const float* be1 = (const float*)d_in[11];
  const float* we2 = (const float*)d_in[12];
  const float* be2 = (const float*)d_in[13];
  const float* wmu = (const float*)d_in[14];
  const float* bmu = (const float*)d_in[15];
  const float* wlv = (const float*)d_in[16];
  const float* blv = (const float*)d_in[17];
  const float* wf  = (const float*)d_in[18];
  const float* bf  = (const float*)d_in[19];
  float* out = (float*)d_out;
  float* mu_out = out + (size_t)2 * 4096 * 512;
  float* lv_out = mu_out + 128;

  char* ws = (char*)d_ws;
  size_t off = 0;
  auto alloc = [&](size_t bytes) -> void* {
    void* p = ws + off; off += (bytes + 255) & ~(size_t)255; return p;
  };
  const size_t NX = (size_t)2 * 4096 * 512;   // 4,194,304
  const size_t NW = (size_t)512 * 512;        // 262,144
  __bf16* xb   = (__bf16*)alloc(NX * 2);
  __bf16* wqb  = (__bf16*)alloc(NW * 2);
  __bf16* wkb  = (__bf16*)alloc(NW * 2);
  __bf16* wvb  = (__bf16*)alloc(NW * 2);
  __bf16* wob  = (__bf16*)alloc(NW * 2);
  __bf16* qg   = (__bf16*)alloc(NX * 2);
  __bf16* kg   = (__bf16*)alloc(NX * 2);
  __bf16* vt   = (__bf16*)alloc(NX * 2);
  __bf16* ctxb = (__bf16*)alloc(NX * 2);
  float*  xm   = (float*)alloc(1024 * 4);
  float*  lat  = (float*)alloc(1024 * 4);

  // 1) conversions
  cvt_bf16<<<(int)((NX + 255) / 256), 256, 0, stream>>>(x, xb, (int)NX);
  cvt_bf16<<<(int)((NW + 255) / 256), 256, 0, stream>>>(wq, wqb, (int)NW);
  cvt_bf16<<<(int)((NW + 255) / 256), 256, 0, stream>>>(wk, wkb, (int)NW);
  cvt_bf16<<<(int)((NW + 255) / 256), 256, 0, stream>>>(wv, wvb, (int)NW);
  cvt_bf16<<<(int)((NW + 255) / 256), 256, 0, stream>>>(wo, wob, (int)NW);
  // 2) VAE branch
  mean_pool<<<2, 512, 0, stream>>>(x, xm);
  vae_mlp<<<1, 512, 0, stream>>>(xm, eps, we1, be1, we2, be2, wmu, bmu,
                                 wlv, blv, wf, bf, mu_out, lv_out, lat);
  // 3) QKV projections + latent guidance (WMMA, 64x64 per wave)
  qkv_gemm<<<dim3(2, 128, 3), dim3(32, 4, 1), 0, stream>>>(
      xb, wqb, wkb, wvb, bq, bk, bv, lat, qg, kg, vt);
  // 4) flash attention (WMMA, 32 Q rows per wave)
  attn_fa<<<512, 128, 0, stream>>>(qg, kg, vt, ctxb);
  // 5) output projection (WMMA, fp32 out)
  out_gemm<<<dim3(2, 128, 1), dim3(32, 4, 1), 0, stream>>>(ctxb, wob, bo, out);
}
